// Attention_35596688949630
// MI455X (gfx1250) — matrix operand order repaired and checked by reading, not compiled, not run on hardware
//
#include <hip/hip_runtime.h>
#include <math.h>

// Problem constants (match reference)
#define B_      4
#define N_      1024
#define DIM_    1024
#define H_      16
#define DH_     64
#define INNER_  1024
#define MAXPOS_ 256
#define TOPK_   64
#define SCALE_  0.125f          // DH^-0.5
#define NEG_    (-3.402823466e38f)

typedef __bf16 bf16;
typedef __attribute__((ext_vector_type(16))) __bf16 v16bf;
typedef __attribute__((ext_vector_type(8)))  __bf16 v8bf;
typedef __attribute__((ext_vector_type(8)))  float  v8f;
typedef __attribute__((ext_vector_type(4)))  unsigned int v4u;
typedef __attribute__((ext_vector_type(8)))  int v8i;
typedef __attribute__((ext_vector_type(4)))  int v4i;

// LDS B-slice row pitch: 64 bf16 payload + 8 bf16 pad = 144B (conflict-free)
#define BPITCH 72

// ---------------------------------------------------------------------------
// WMMA helpers (CDNA5 v_wmma_f32_16x16x32_bf16, wave32)
// A tile (16x32, MxK): lane m=L%16, half=L/16; elem e -> K=16*half+e
//                      (same K order as B)
// B tile fed as Bt (NxK row-major): lane n=L%16, half=L/16; elem e -> K=16*half+e
// C/D f32: lane n=L%16; reg r -> row m=r+8*(L/16)
// ---------------------------------------------------------------------------
__device__ __forceinline__ v16bf load_a_tile(const bf16* __restrict__ base, int ld) {
  const int lane = threadIdx.x & 31;
  const int m = lane & 15, h = lane >> 4;
  const bf16* p = base + (size_t)m * ld + 16 * h;
  v8bf lo = *(const v8bf*)(p);
  v8bf hi = *(const v8bf*)(p + 8);
  v16bf r;
#pragma unroll
  for (int i = 0; i < 8; ++i) { r[i] = lo[i]; r[i + 8] = hi[i]; }
  return r;
}

__device__ __forceinline__ v16bf load_bt_tile(const bf16* __restrict__ base, int ld) {
  const int lane = threadIdx.x & 31;
  const int n = lane & 15, h = lane >> 4;
  const bf16* p = base + (size_t)n * ld + 16 * h;
  v8bf lo = *(const v8bf*)(p);
  v8bf hi = *(const v8bf*)(p + 8);
  v16bf r;
#pragma unroll
  for (int i = 0; i < 8; ++i) { r[i] = lo[i]; r[i + 8] = hi[i]; }
  return r;
}

// B tile read from LDS slice (row pitch BPITCH elements)
__device__ __forceinline__ v16bf load_bt_lds(const bf16* base, int row0, int ks) {
  const int lane = threadIdx.x & 31;
  const int n = lane & 15, h = lane >> 4;
  const bf16* p = base + (size_t)(row0 + n) * BPITCH + ks + 16 * h;
  v8bf lo = *(const v8bf*)(p);
  v8bf hi = *(const v8bf*)(p + 8);
  v16bf r;
#pragma unroll
  for (int i = 0; i < 8; ++i) { r[i] = lo[i]; r[i + 8] = hi[i]; }
  return r;
}

__device__ __forceinline__ v8f wmma_bf16(v16bf a, v16bf b, v8f c) {
  return __builtin_amdgcn_wmma_f32_16x16x32_bf16(false, a, false, b, (short)0, c,
                                                 false, false);
}

__device__ __forceinline__ void zero_acc(v8f* acc) {
#pragma unroll
  for (int t = 0; t < 4; ++t)
#pragma unroll
    for (int i = 0; i < 8; ++i) acc[t][i] = 0.f;
}

// ---------------------------------------------------------------------------
// Tensor Data Mover: async-load a 64(rows) x 64(K) bf16 tile (row stride 1024
// elements in memory) into LDS at lds_off with 16B padding every 128B row,
// giving BPITCH-element LDS rows.  D# per CDNA5 ISA.
// Toolchain uses the 6-arg builtin: (v4u, v8i, v4i, v4i, v8i, i32 cpol).
// ---------------------------------------------------------------------------
__device__ __forceinline__ void tdm_load_tile(unsigned lds_off, const bf16* gptr) {
  const unsigned long long ga = (unsigned long long)(uintptr_t)gptr;
  v4u g0;
  g0[0] = 1u;                                       // count=1, user descriptor
  g0[1] = lds_off;                                  // LDS byte address
  g0[2] = (unsigned)(ga & 0xFFFFFFFFu);             // global_addr[31:0]
  g0[3] = (unsigned)((ga >> 32) & 0x01FFFFFFu) | (2u << 30);  // addr[56:32]|type=2
  v8i g1;
  // data_size=2B(1), pad_enable, pad_interval=32DW(code4), pad_amount=4DW(code3)
  g1[0] = (int)((1u << 16) | (1u << 20) | (4u << 22) | (3u << 25));
  g1[1] = (int)(1024u << 16);                       // tensor_dim0 lo16 = 1024
  g1[2] = (int)(1024u << 16);                       // dim0 hi=0 | tensor_dim1 lo16=1024
  g1[3] = (int)(64u << 16);                         // dim1 hi=0 | tile_dim0=64
  g1[4] = (int)(64u);                               // tile_dim1=64, tile_dim2=0
  g1[5] = (int)(1024u);                             // tensor_dim0_stride lo32 = 1024
  g1[6] = 0;                                        // stride hi | dim1_stride lo
  g1[7] = 0;
  const v4i z4 = {0, 0, 0, 0};                      // groups 2/3 unused (2D tile)
  const v8i z8 = {0, 0, 0, 0, 0, 0, 0, 0};
  __builtin_amdgcn_tensor_load_to_lds(g0, g1, z4, z4, z8, 0);
}

// ---------------------------------------------------------------------------
// Conversion / transpose kernels (one-time, small)
// ---------------------------------------------------------------------------
__global__ void k_cvt(const float* __restrict__ s, bf16* __restrict__ d, size_t n) {
  size_t i = (size_t)blockIdx.x * blockDim.x + threadIdx.x;
  const size_t stride = (size_t)gridDim.x * blockDim.x;
  for (; i < n; i += stride) d[i] = (bf16)s[i];
}

// dst[c*R + r] = src[r*C + c]  (f32 RxC -> bf16 CxR)
__global__ void k_transpose_cvt(const float* __restrict__ s, bf16* __restrict__ d,
                                int R, int C) {
  size_t i = (size_t)blockIdx.x * blockDim.x + threadIdx.x;
  const size_t n = (size_t)R * C;
  const size_t stride = (size_t)gridDim.x * blockDim.x;
  for (; i < n; i += stride) {
    const int r = (int)(i / C), c = (int)(i % C);
    d[(size_t)c * R + r] = (bf16)s[i];
  }
}

// ---------------------------------------------------------------------------
// TDM-staged GEMM: C(4096x1024) = X(4096x1024,bf16) @ W + bias.
// Block = 8 waves -> 128 rows x 64 cols; the 64x1024 weight slice is shared by
// all waves and streamed through LDS in 64-K chunks by the Tensor Data Mover
// (double buffered, wave0 is producer).  A operand streams from global.
// mode 0: dst f32 row-major (final projection)
// mode 1: dst bf16 head-blocked [b,h,n,d]      (q, k)
// mode 2: dst bf16 head-blocked [b,h,d,n]      (v^T)
// Grid: 32 m-blocks x 16 col-groups = 512 blocks.
// ---------------------------------------------------------------------------
__global__ void __launch_bounds__(256)
k_gemm_tdm(const bf16* __restrict__ X, const bf16* __restrict__ WT,
           const float* __restrict__ bias, void* __restrict__ dst, int mode) {
  __shared__ __align__(16) bf16 sB[2][64 * BPITCH];
  const int wid = threadIdx.x >> 5;
  const int lane = threadIdx.x & 31;
  const int mb = blockIdx.x >> 4;
  const int cg = blockIdx.x & 15;
  const int m0 = mb * 128 + wid * 16;
  const int c0 = cg * 64;
  const bf16* Arow = X + (size_t)m0 * DIM_;
  const bf16* Brow = WT + (size_t)c0 * DIM_;     // 64 rows x 1024 K
  const unsigned lds0 = (unsigned)(uintptr_t)&sB[0][0];
  const unsigned lds1 = (unsigned)(uintptr_t)&sB[1][0];

  v8f acc[4]; zero_acc(acc);

  if (wid == 0) tdm_load_tile(lds0, Brow);       // prologue: chunk 0 -> buf0
  for (int kc = 0; kc < 16; ++kc) {
    if (wid == 0) {
      if (kc < 15) {
        tdm_load_tile((kc & 1) ? lds0 : lds1, Brow + (kc + 1) * 64);
        __builtin_amdgcn_s_wait_tensorcnt(1);    // chunk kc complete
      } else {
        __builtin_amdgcn_s_wait_tensorcnt(0);
      }
    }
    __syncthreads();                             // buf[kc&1] ready for all waves
    const bf16* bufp = (kc & 1) ? &sB[1][0] : &sB[0][0];
    const int kbase = kc * 64;
    __builtin_prefetch(Arow + kbase + 128, 0, 0);  // stream A ahead into L2
#pragma unroll
    for (int ks = 0; ks < 64; ks += 32) {
      v16bf a = load_a_tile(Arow + kbase + ks, DIM_);
#pragma unroll
      for (int t = 0; t < 4; ++t) {
        v16bf b = load_bt_lds(bufp, 16 * t, ks);
        acc[t] = wmma_bf16(a, b, acc[t]);
      }
    }
    __syncthreads();                             // before buffer reuse
  }

  const int half = lane >> 4, nl = lane & 15;
#pragma unroll
  for (int t = 0; t < 4; ++t) {
    const int c = c0 + 16 * t + nl;
    const float bv = bias[c];
#pragma unroll
    for (int r = 0; r < 8; ++r) {
      const int m = m0 + r + 8 * half;
      const float v = acc[t][r] + bv;
      if (mode == 0) {
        ((float*)dst)[(size_t)m * DIM_ + c] = v;
      } else {
        const int b = m >> 10, n = m & 1023;
        const int hh = c >> 6, d = c & 63;
        const size_t idx = (mode == 2)
            ? (((size_t)(b * H_ + hh) * DH_ + d) * N_ + n)
            : (((size_t)(b * H_ + hh) * N_ + n) * DH_ + d);
        ((bf16*)dst)[idx] = (bf16)v;
      }
    }
  }
}

// ---------------------------------------------------------------------------
// relscore[bh,n,p] = q[bh,n,:] . rel_emb[p,:]  (K=64: all loads first, then 8 WMMAs)
// ---------------------------------------------------------------------------
__global__ void __launch_bounds__(256)
k_relscore(const bf16* __restrict__ Q, const bf16* __restrict__ REL,
           float* __restrict__ RS) {
  const int w = blockIdx.x * 8 + (threadIdx.x >> 5);
  const int lane = threadIdx.x & 31;
  const int bh = w >> 8, rem = w & 255;
  const int m0 = (rem >> 2) * 16, c0 = (rem & 3) * 64;
  const bf16* A = Q + (size_t)bh * N_ * DH_ + (size_t)m0 * DH_;
  v8f acc[4]; zero_acc(acc);
  v16bf a0 = load_a_tile(A, DH_);
  v16bf a1 = load_a_tile(A + 32, DH_);
  v16bf b0[4], b1[4];
#pragma unroll
  for (int t = 0; t < 4; ++t) {
    b0[t] = load_bt_tile(REL + (size_t)(c0 + 16 * t) * DH_, DH_);
    b1[t] = load_bt_tile(REL + (size_t)(c0 + 16 * t) * DH_ + 32, DH_);
  }
#pragma unroll
  for (int t = 0; t < 4; ++t) acc[t] = wmma_bf16(a0, b0[t], acc[t]);
#pragma unroll
  for (int t = 0; t < 4; ++t) acc[t] = wmma_bf16(a1, b1[t], acc[t]);

  const int half = lane >> 4, nl = lane & 15;
#pragma unroll
  for (int t = 0; t < 4; ++t) {
    const int c = c0 + 16 * t + nl;
#pragma unroll
    for (int r = 0; r < 8; ++r) {
      const int m = m0 + r + 8 * half;
      RS[((size_t)bh * N_ + m) * MAXPOS_ + c] = acc[t][r];
    }
  }
}

// ---------------------------------------------------------------------------
// dots[bh,i,j] = (q.k + relscore[bh,i,dist(i,j)]) * SCALE, masked.
// K=64: all 20 b128 loads issued up front, then 8 WMMAs.
// ---------------------------------------------------------------------------
__global__ void __launch_bounds__(256)
k_dots(const bf16* __restrict__ Q, const bf16* __restrict__ K,
       const float* __restrict__ RS, const unsigned char* __restrict__ mask,
       float* __restrict__ D) {
  const int w = blockIdx.x * 8 + (threadIdx.x >> 5);
  const int lane = threadIdx.x & 31;
  const int bh = w >> 10, rem = w & 1023;
  const int m0 = (rem >> 4) * 16, c0 = (rem & 15) * 64;
  const int b = bh >> 4;
  const bf16* A  = Q + (size_t)bh * N_ * DH_ + (size_t)m0 * DH_;
  const bf16* Bt = K + (size_t)bh * N_ * DH_;
  v8f acc[4]; zero_acc(acc);
  v16bf a0 = load_a_tile(A, DH_);
  v16bf a1 = load_a_tile(A + 32, DH_);
  v16bf b0[4], b1[4];
#pragma unroll
  for (int t = 0; t < 4; ++t) {
    b0[t] = load_bt_tile(Bt + (size_t)(c0 + 16 * t) * DH_, DH_);
    b1[t] = load_bt_tile(Bt + (size_t)(c0 + 16 * t) * DH_ + 32, DH_);
  }
#pragma unroll
  for (int t = 0; t < 4; ++t) acc[t] = wmma_bf16(a0, b0[t], acc[t]);
#pragma unroll
  for (int t = 0; t < 4; ++t) acc[t] = wmma_bf16(a1, b1[t], acc[t]);

  const int half = lane >> 4, nl = lane & 15;
#pragma unroll
  for (int t = 0; t < 4; ++t) {
    const int j = c0 + 16 * t + nl;
    const bool mj = mask[b * N_ + j] != 0;
#pragma unroll
    for (int r = 0; r < 8; ++r) {
      const int i = m0 + r + 8 * half;
      int dist = i - j + MAXPOS_;
      dist = dist < 0 ? 0 : (dist > MAXPOS_ - 1 ? MAXPOS_ - 1 : dist);
      float v = (acc[t][r] + RS[((size_t)bh * N_ + i) * MAXPOS_ + dist]) * SCALE_;
      if (!(mj && (mask[b * N_ + i] != 0))) v = NEG_;
      D[((size_t)bh * N_ + i) * N_ + j] = v;
    }
  }
}

// ---------------------------------------------------------------------------
// Per-row exact top-64 (32-bit MSB radix-select on monotonic keys) + softmax
// * gating.  attn (bf16) written IN PLACE into the dots buffer (row fully in
// registers before overwrite -> race-free).  One 256-thread block per row.
// ---------------------------------------------------------------------------
__global__ void __launch_bounds__(256)
k_topk_softmax(const float* __restrict__ D, const float* __restrict__ G,
               bf16* __restrict__ ATT) {
  const int row = blockIdx.x;                 // bh*N + i
  const int tid = threadIdx.x;
  const float* dr = D + (size_t)row * N_;

  float val[4]; unsigned key[4];
#pragma unroll
  for (int ii = 0; ii < 4; ++ii) {
    const int j = tid + 256 * ii;
    const float f = dr[j];
    const unsigned u = __float_as_uint(f);
    key[ii] = (u & 0x80000000u) ? ~u : (u | 0x80000000u);  // order-preserving
    val[ii] = f;
  }

  __shared__ unsigned s_cnt;
  __shared__ unsigned s_max;
  __shared__ float    s_sum;

  if (tid == 0) s_max = 0u;
  __syncthreads();
  unsigned mymax = key[0];
#pragma unroll
  for (int ii = 1; ii < 4; ++ii) mymax = key[ii] > mymax ? key[ii] : mymax;
  atomicMax(&s_max, mymax);
  __syncthreads();
  const unsigned maxkey = s_max;

  unsigned prefix = 0u;
  int krem = TOPK_;
  for (int bpos = 31; bpos >= 0; --bpos) {
    const unsigned bit = 1u << bpos;
    const unsigned hi = prefix | bit;
    const unsigned hm = ~(bit - 1u);            // bits >= bpos
    if (tid == 0) s_cnt = 0u;
    __syncthreads();
    unsigned c = 0;
#pragma unroll
    for (int ii = 0; ii < 4; ++ii) c += ((key[ii] & hm) == hi) ? 1u : 0u;
    if (c) atomicAdd(&s_cnt, c);
    __syncthreads();
    const unsigned total = s_cnt;
    __syncthreads();                            // protect s_cnt reuse
    if ((int)total >= krem) prefix = hi; else krem -= (int)total;
  }
  const unsigned T = prefix;

  const unsigned mu = (maxkey & 0x80000000u) ? (maxkey & 0x7fffffffu) : ~maxkey;
  const float fmax = __uint_as_float(mu);

  if (tid == 0) s_sum = 0.f;
  __syncthreads();
  float e[4]; float s = 0.f;
#pragma unroll
  for (int ii = 0; ii < 4; ++ii) {
    e[ii] = (key[ii] >= T) ? __expf(val[ii] - fmax) : 0.f;
    s += e[ii];
  }
  atomicAdd(&s_sum, s);
  __syncthreads();
  const float inv = 1.f / s_sum;

  const float* gr = G + (size_t)row * N_;
#pragma unroll
  for (int ii = 0; ii < 4; ++ii) {
    const int j = tid + 256 * ii;
    ATT[(size_t)row * (2 * N_) + j] = (bf16)(e[ii] * inv * gr[j]);
  }
}

// ---------------------------------------------------------------------------
// inner[b,n,h*64+d] = attn[bh,n,:] @ v[bh,:,d]   (attn lda=2048, v as v^T).
// Register double-buffered K loop (unroll x2) so WMMAs overlap loads.
// ---------------------------------------------------------------------------
__global__ void __launch_bounds__(256)
k_av(const bf16* __restrict__ ATT, const bf16* __restrict__ VT,
     bf16* __restrict__ OUT) {
  const int w = blockIdx.x * 8 + (threadIdx.x >> 5);
  const int lane = threadIdx.x & 31;
  const int bh = w >> 6, m0 = (w & 63) * 16;
  const int b = bh >> 4, h = bh & 15;
  const bf16* A  = ATT + ((size_t)bh * N_ + m0) * (2 * N_);
  const bf16* Bt = VT + (size_t)bh * DH_ * N_;
  v8f acc[4]; zero_acc(acc);

  v16bf aA = load_a_tile(A, 2 * N_);
  v16bf bA[4], bB[4];
#pragma unroll
  for (int t = 0; t < 4; ++t) bA[t] = load_bt_tile(Bt + (size_t)(16 * t) * N_, N_);

  for (int k0 = 0; k0 < N_; k0 += 64) {
    __builtin_prefetch(A + k0 + 256, 0, 0);
    v16bf aB = load_a_tile(A + k0 + 32, 2 * N_);
#pragma unroll
    for (int t = 0; t < 4; ++t)
      bB[t] = load_bt_tile(Bt + (size_t)(16 * t) * N_ + k0 + 32, N_);
#pragma unroll
    for (int t = 0; t < 4; ++t) acc[t] = wmma_bf16(aA, bA[t], acc[t]);
    if (k0 + 64 < N_) {
      aA = load_a_tile(A + k0 + 64, 2 * N_);
#pragma unroll
      for (int t = 0; t < 4; ++t)
        bA[t] = load_bt_tile(Bt + (size_t)(16 * t) * N_ + k0 + 64, N_);
    }
#pragma unroll
    for (int t = 0; t < 4; ++t) acc[t] = wmma_bf16(aB, bB[t], acc[t]);
  }

  const int half = lane >> 4, nl = lane & 15;
#pragma unroll
  for (int t = 0; t < 4; ++t) {
    const int d = 16 * t + nl;
#pragma unroll
    for (int r = 0; r < 8; ++r) {
      const int n = m0 + r + 8 * half;
      OUT[((size_t)b * N_ + n) * INNER_ + h * DH_ + d] = (bf16)acc[t][r];
    }
  }
}

// ---------------------------------------------------------------------------
extern "C" void kernel_launch(void* const* d_in, const int* in_sizes, int n_in,
                              void* d_out, int out_size, void* d_ws, size_t ws_size,
                              hipStream_t stream) {
  const float*         x      = (const float*)d_in[0];
  const unsigned char* mask   = (const unsigned char*)d_in[1];
  const float*         gating = (const float*)d_in[2];
  const float*         Wq     = (const float*)d_in[3];
  const float*         bq     = (const float*)d_in[4];
  const float*         Wkv    = (const float*)d_in[5];
  const float*         bkv    = (const float*)d_in[6];
  const float*         Wo     = (const float*)d_in[7];
  const float*         bo     = (const float*)d_in[8];
  const float*         rel    = (const float*)d_in[9];
  float*               out    = (float*)d_out;

  char* ws = (char*)d_ws;
  size_t off = 0;
  auto alloc = [&](size_t bytes) -> char* {
    char* p = ws + off;
    off = (off + bytes + 255) & ~(size_t)255;
    return p;
  };

  bf16*  xb     = (bf16*)alloc((size_t)B_ * N_ * DIM_ * 2);
  bf16*  WqT    = (bf16*)alloc((size_t)INNER_ * DIM_ * 2);
  bf16*  WkvT   = (bf16*)alloc((size_t)2 * INNER_ * DIM_ * 2);
  bf16*  WoT    = (bf16*)alloc((size_t)DIM_ * INNER_ * 2);
  bf16*  relb   = (bf16*)alloc((size_t)MAXPOS_ * DH_ * 2);
  bf16*  qb     = (bf16*)alloc((size_t)B_ * H_ * N_ * DH_ * 2);
  bf16*  kb     = (bf16*)alloc((size_t)B_ * H_ * N_ * DH_ * 2);
  bf16*  vT     = (bf16*)alloc((size_t)B_ * H_ * DH_ * N_ * 2);
  bf16*  innerb = (bf16*)alloc((size_t)B_ * N_ * INNER_ * 2);
  float* RS     = (float*)alloc((size_t)B_ * H_ * N_ * MAXPOS_ * 4);
  float* DOTS   = (float*)alloc((size_t)B_ * H_ * N_ * N_ * 4);   // also holds bf16 attn

  // 1) precision conversion + weight transposes (contiguous WMMA B-loads)
  k_cvt<<<2048, 256, 0, stream>>>(x, xb, (size_t)B_ * N_ * DIM_);
  k_transpose_cvt<<<2048, 256, 0, stream>>>(Wq, WqT, DIM_, INNER_);
  k_transpose_cvt<<<4096, 256, 0, stream>>>(Wkv, WkvT, DIM_, 2 * INNER_);
  k_transpose_cvt<<<2048, 256, 0, stream>>>(Wo, WoT, INNER_, DIM_);
  k_cvt<<<64, 256, 0, stream>>>(rel, relb, (size_t)MAXPOS_ * DH_);

  // 2) q / k / v projections (TDM-staged weights; v stored transposed per head)
  k_gemm_tdm<<<512, 256, 0, stream>>>(xb, WqT, bq, qb, 1);
  k_gemm_tdm<<<512, 256, 0, stream>>>(xb, WkvT, bkv, kb, 1);
  k_gemm_tdm<<<512, 256, 0, stream>>>(xb, WkvT + (size_t)INNER_ * DIM_,
                                      bkv + INNER_, vT, 2);

  // 3) relative-position scores: (bh,n,256) = q @ rel_emb^T
  k_relscore<<<2048, 256, 0, stream>>>(qb, relb, RS);

  // 4) dots = (q.k^T + rel gather) * scale, masked
  k_dots<<<8192, 256, 0, stream>>>(qb, kb, RS, mask, DOTS);

  // 5) per-row exact top-64 + softmax * gating -> bf16 attn (in-place)
  k_topk_softmax<<<B_ * H_ * N_, 256, 0, stream>>>(DOTS, gating, (bf16*)DOTS);

  // 6) attn @ v -> inner (bf16, (b,n,h*dh))
  k_av<<<512, 256, 0, stream>>>((const bf16*)DOTS, vT, innerb);

  // 7) inner @ Wo + bo -> out (f32, TDM-staged weights)
  k_gemm_tdm<<<512, 256, 0, stream>>>(innerb, WoT, bo, out, 0);
}
